// SelectiveSSM_37237366456735
// MI455X (gfx1250) — matrix operand order repaired and checked by reading, not compiled, not run on hardware
//
#include <hip/hip_runtime.h>
#include <hip/hip_bf16.h>
#include <math.h>

#define DIMD   2048
#define SEQ    2048
#define NBATCH 4
#define NSTATE 64
#define DTRANK 128
#define CHUNKL 512
#define BLTOK  (NBATCH * SEQ)   // 8192 tokens

typedef __attribute__((ext_vector_type(16))) __bf16 v16bf;
typedef __attribute__((ext_vector_type(8)))  __bf16 v8bf;
typedef __attribute__((ext_vector_type(8)))  float  v8f;
typedef __attribute__((ext_vector_type(4)))  unsigned int v4u;
typedef __attribute__((ext_vector_type(8)))  int v8i;
typedef __attribute__((ext_vector_type(4)))  int v4i;

#if __has_builtin(__builtin_amdgcn_tensor_load_to_lds) && __has_builtin(__builtin_amdgcn_s_wait_tensorcnt)
#define USE_TDM 1
#else
#define USE_TDM 0
#endif

// ---------------------------------------------------------------------------
// Fragment builders (CDNA5 ISA 7.12.2 layouts, wave32). Source rows are
// 32 bf16 elements (64B) whether in LDS tile or global row-major.
// A 16x32 bf16: lane l -> row l%16, K = (l/16)*8 + {0..7} and 16+(l/16)*8+{0..7}
// B 32x16 bf16 (from row-major W(N,K)): lane l -> col l%16, K = (l/16)*8 + {0..7} and 16+(l/16)*8+{0..7}
// ---------------------------------------------------------------------------
__device__ __forceinline__ v16bf make_a_frag(const __bf16* rowp, int hf) {
  const v8bf lo = *(const v8bf*)(rowp + hf * 8);
  const v8bf hi = *(const v8bf*)(rowp + 16 + hf * 8);
  return __builtin_shufflevector(lo, hi, 0,1,2,3,4,5,6,7,8,9,10,11,12,13,14,15);
}
__device__ __forceinline__ v16bf make_b_frag(const __bf16* rowp, int hf) {
  const v8bf lo = *(const v8bf*)(rowp + hf * 8);
  const v8bf hi = *(const v8bf*)(rowp + 16 + hf * 8);
  return __builtin_shufflevector(lo, hi, 0,1,2,3,4,5,6,7,8,9,10,11,12,13,14,15);
}

#if USE_TDM
// ---------------------------------------------------------------------------
// Issue a TDM 2-D tile load: tile_rows x 32 bf16 elements from a row-major
// matrix (row stride = ld elems) into a contiguous LDS tile (64B per row).
// Descriptor per CDNA5 ISA ch.8. All dwords forced uniform via readfirstlane.
// Toolchain ships the 6-arg builtin: (g0 u32x4, g1 i32x8, i32x4, i32x4,
// i32x8, cpol).
// ---------------------------------------------------------------------------
__device__ __forceinline__ void tdm_load_tile(unsigned lds_off, const __bf16* gsrc,
                                              unsigned tile_rows, unsigned ld) {
  const unsigned long long ga = (unsigned long long)(uintptr_t)gsrc;
  const unsigned td0 = ld;              // tensor dim0 (elems)
  const unsigned td1 = 0x00100000u;     // tensor dim1: large, tiles never OOB
  const unsigned long long s0 = ld;     // dim0 stride (elems)

  v4u g0;
  g0[0] = (unsigned)__builtin_amdgcn_readfirstlane(1);               // count=1
  g0[1] = (unsigned)__builtin_amdgcn_readfirstlane((int)lds_off);    // lds_addr
  g0[2] = (unsigned)__builtin_amdgcn_readfirstlane((int)(ga & 0xffffffffu));
  g0[3] = (unsigned)__builtin_amdgcn_readfirstlane(
              (int)(((ga >> 32) & 0x1ffffffu) | (2u << 30)));        // addr hi | type=2

  v8i g1;
  g1[0] = __builtin_amdgcn_readfirstlane((int)(1u << 16));           // data_size=2B
  g1[1] = __builtin_amdgcn_readfirstlane((int)((td0 & 0xffffu) << 16));
  g1[2] = __builtin_amdgcn_readfirstlane((int)(((td0 >> 16) & 0xffffu) |
                                               ((td1 & 0xffffu) << 16)));
  g1[3] = __builtin_amdgcn_readfirstlane((int)(((td1 >> 16) & 0xffffu) |
                                               (32u << 16)));        // tile_dim0=32
  g1[4] = __builtin_amdgcn_readfirstlane((int)(tile_rows & 0xffffu)); // tile_dim1
  g1[5] = __builtin_amdgcn_readfirstlane((int)(s0 & 0xffffffffu));   // dim0 stride lo
  g1[6] = __builtin_amdgcn_readfirstlane((int)((s0 >> 32) & 0xffffu));
  g1[7] = __builtin_amdgcn_readfirstlane(0);

  const v4i gz4 = {0, 0, 0, 0};                       // 2-D tensor: groups 2/3 null
  const v8i gz8 = {0, 0, 0, 0, 0, 0, 0, 0};
  __builtin_amdgcn_tensor_load_to_lds(g0, g1, gz4, gz4, gz8, 0);
}
#endif

// ---------------------------------------------------------------------------
// GEMM: out[M,N] = act( A[M,K](bf16) * W[N,K]^T(bf16) + bias[N] ), f32 accum.
// Block = 256 threads = 8 waves; wave tile 32x32 (2x2 WMMA), block tile 128x64.
// TDM double-buffers 128x32 A-tiles and 64x32 W-tiles through LDS.
// M%128==0, N%64==0, K%32==0 guaranteed by launches.
// ---------------------------------------------------------------------------
__global__ __launch_bounds__(256)
void gemm_bf16_wmma(const __bf16* __restrict__ A, int lda,
                    const __bf16* __restrict__ W, int ldw,
                    const float* __restrict__ bias,
                    float* __restrict__ out, int ldo,
                    int K, int act) {
  const int lane = threadIdx.x & 31;
  const int wave = threadIdx.x >> 5;
  const int wm   = wave & 3;        // 4 waves along M
  const int wn   = wave >> 2;       // 2 waves along N
  const int hf   = lane >> 4;       // lane half-group
  const int lr   = lane & 15;

  const int bm = blockIdx.x * 128;
  const int bn = blockIdx.y * 64;
  const int m0 = bm + wm * 32;
  const int n0 = bn + wn * 32;

  v8f c00 = {}, c01 = {}, c10 = {}, c11 = {};

#if USE_TDM
  __shared__ __bf16 As[2][128][32];   // 16 KB
  __shared__ __bf16 Ws[2][64][32];    //  8 KB

  if (threadIdx.x == 0) {
    tdm_load_tile((unsigned)(uintptr_t)&As[0][0][0], A + (size_t)bm * lda, 128, (unsigned)lda);
    tdm_load_tile((unsigned)(uintptr_t)&Ws[0][0][0], W + (size_t)bn * ldw, 64,  (unsigned)ldw);
  }

  int buf = 0;
  for (int k = 0; k < K; k += 32, buf ^= 1) {
    if (wave == 0) __builtin_amdgcn_s_wait_tensorcnt(0);  // current tiles landed
    __syncthreads();
    if (threadIdx.x == 0 && (k + 32) < K) {               // prefetch next tiles via TDM
      tdm_load_tile((unsigned)(uintptr_t)&As[buf ^ 1][0][0],
                    A + (size_t)bm * lda + (k + 32), 128, (unsigned)lda);
      tdm_load_tile((unsigned)(uintptr_t)&Ws[buf ^ 1][0][0],
                    W + (size_t)bn * ldw + (k + 32), 64, (unsigned)ldw);
    }
    const v16bf a0 = make_a_frag(&As[buf][wm * 32 + lr][0],      hf);
    const v16bf a1 = make_a_frag(&As[buf][wm * 32 + 16 + lr][0], hf);
    const v16bf b0 = make_b_frag(&Ws[buf][wn * 32 + lr][0],      hf);
    const v16bf b1 = make_b_frag(&Ws[buf][wn * 32 + 16 + lr][0], hf);

    c00 = __builtin_amdgcn_wmma_f32_16x16x32_bf16(false, a0, false, b0, (short)0, c00, false, false);
    c01 = __builtin_amdgcn_wmma_f32_16x16x32_bf16(false, a0, false, b1, (short)0, c01, false, false);
    c10 = __builtin_amdgcn_wmma_f32_16x16x32_bf16(false, a1, false, b0, (short)0, c10, false, false);
    c11 = __builtin_amdgcn_wmma_f32_16x16x32_bf16(false, a1, false, b1, (short)0, c11, false, false);
    // No trailing barrier needed: each wave's ds reads complete before its
    // WMMAs, which precede the next iteration's barrier in program order.
  }
#else
  for (int k = 0; k < K; k += 32) {
    const v16bf a0 = make_a_frag(A + (size_t)(m0 + lr) * lda + k,      hf);
    const v16bf a1 = make_a_frag(A + (size_t)(m0 + 16 + lr) * lda + k, hf);
    const v16bf b0 = make_b_frag(W + (size_t)(n0 + lr) * ldw + k,      hf);
    const v16bf b1 = make_b_frag(W + (size_t)(n0 + 16 + lr) * ldw + k, hf);
    if (k + 32 < K) {
      __builtin_prefetch(A + (size_t)(m0 + lr) * lda + k + 32, 0, 1);
      __builtin_prefetch(W + (size_t)(n0 + lr) * ldw + k + 32, 0, 1);
    }
    c00 = __builtin_amdgcn_wmma_f32_16x16x32_bf16(false, a0, false, b0, (short)0, c00, false, false);
    c01 = __builtin_amdgcn_wmma_f32_16x16x32_bf16(false, a0, false, b1, (short)0, c01, false, false);
    c10 = __builtin_amdgcn_wmma_f32_16x16x32_bf16(false, a1, false, b0, (short)0, c10, false, false);
    c11 = __builtin_amdgcn_wmma_f32_16x16x32_bf16(false, a1, false, b1, (short)0, c11, false, false);
  }
#endif

  // Epilogue. C/D layout: VGPR r -> row r + 8*hf, col = lr.
  auto store_tile = [&](const v8f& c, int mt, int nt) {
    const int col = n0 + nt * 16 + lr;
    const float bv = bias ? bias[col] : 0.0f;
#pragma unroll
    for (int r = 0; r < 8; ++r) {
      const int row = m0 + mt * 16 + hf * 8 + r;
      float v = c[r] + bv;
      if (act) v = (v > 20.0f) ? v : log1pf(__expf(v));   // softplus
      out[(size_t)row * ldo + col] = v;
    }
  };
  store_tile(c00, 0, 0); store_tile(c01, 0, 1);
  store_tile(c10, 1, 0); store_tile(c11, 1, 1);
}

// ---------------------------------------------------------------------------
// LayerNorm over D per token; writes f32 (scan input) and bf16 (GEMM input).
// ---------------------------------------------------------------------------
__global__ __launch_bounds__(256)
void layernorm_kernel(const float* __restrict__ x, const float* __restrict__ w,
                      const float* __restrict__ bvec,
                      float* __restrict__ xn, __bf16* __restrict__ xnb) {
  const size_t base = (size_t)blockIdx.x * DIMD;
  __shared__ float s_sum[256], s_sq[256];
  float s = 0.f, ss = 0.f;
  for (int i = threadIdx.x; i < DIMD; i += 256) {
    const float v = x[base + i];
    s += v; ss += v * v;
  }
  s_sum[threadIdx.x] = s; s_sq[threadIdx.x] = ss;
  __syncthreads();
  for (int off = 128; off > 0; off >>= 1) {
    if ((int)threadIdx.x < off) {
      s_sum[threadIdx.x] += s_sum[threadIdx.x + off];
      s_sq[threadIdx.x]  += s_sq[threadIdx.x + off];
    }
    __syncthreads();
  }
  const float mu   = s_sum[0] * (1.0f / DIMD);
  const float var  = s_sq[0] * (1.0f / DIMD) - mu * mu;
  const float rstd = rsqrtf(var + 1e-5f);
  for (int i = threadIdx.x; i < DIMD; i += 256) {
    const float v = (x[base + i] - mu) * rstd * w[i] + bvec[i];
    xn[base + i]  = v;
    xnb[base + i] = (__bf16)v;
  }
}

__global__ void f32_to_bf16_kernel(const float* __restrict__ in,
                                   __bf16* __restrict__ out, size_t n) {
  size_t i = (size_t)blockIdx.x * blockDim.x + threadIdx.x;
  const size_t stride = (size_t)gridDim.x * blockDim.x;
  for (; i < n; i += stride) out[i] = (__bf16)in[i];
}

// delta_in[t, 0:128] = bf16(proj[t, 0:128])
__global__ void slice_delta_bf16_kernel(const float* __restrict__ proj,
                                        __bf16* __restrict__ out) {
  size_t i = (size_t)blockIdx.x * blockDim.x + threadIdx.x;
  const size_t stride = (size_t)gridDim.x * blockDim.x;
  const size_t total = (size_t)BLTOK * DTRANK;
  for (; i < total; i += stride) {
    const size_t row = i >> 7, col = i & 127;
    out[i] = (__bf16)proj[row * 256 + col];
  }
}

// ---------------------------------------------------------------------------
// Selective scan: one thread per (b,d), h[64] + A-row in registers,
// B/C rows broadcast through LDS. Emits y2 = y + x*D_param in bf16.
// proj row layout: [ delta(128) | B(64) | C(64) ]; C row index wraps at 512.
// ---------------------------------------------------------------------------
__global__ __launch_bounds__(256)
void scan_kernel(const float* __restrict__ xn, const float* __restrict__ proj,
                 const float* __restrict__ delta, const float* __restrict__ A_log,
                 const float* __restrict__ Dp, const float* __restrict__ x,
                 __bf16* __restrict__ y2) {
  const int b = blockIdx.y;
  const int d = blockIdx.x * 256 + threadIdx.x;

  float a[NSTATE], h[NSTATE];
#pragma unroll
  for (int n = 0; n < NSTATE; ++n) {
    a[n] = -__expf(A_log[(size_t)d * NSTATE + n]);  // A = -exp(A_log)
    h[n] = 0.0f;
  }
  const float dpd = Dp[d];

  __shared__ float Bs[NSTATE], Cs[NSTATE];

  for (int t = 0; t < SEQ; ++t) {
    __syncthreads();  // protect prior-iteration LDS reads
    if ((int)threadIdx.x < NSTATE) {
      Bs[threadIdx.x] = proj[((size_t)b * SEQ + t) * 256 + DTRANK + threadIdx.x];
    } else if ((int)threadIdx.x < 2 * NSTATE) {
      const int n = threadIdx.x - NSTATE;
      Cs[n] = proj[((size_t)b * SEQ + (t & (CHUNKL - 1))) * 256 + DTRANK + NSTATE + n];
    }
    __syncthreads();

    const size_t off = ((size_t)b * SEQ + t) * DIMD + d;
    const float dt  = delta[off];
    const float dtx = dt * xn[off];
    float y = 0.0f;
#pragma unroll
    for (int n = 0; n < NSTATE; ++n) {
      const float dA = __expf(dt * a[n]);           // v_exp_f32 (TRANS)
      h[n] = fmaf(dA, h[n], dtx * Bs[n]);
      y    = fmaf(h[n], Cs[n], y);
    }
    y2[off] = (__bf16)(y + x[off] * dpd);
  }
}

// ---------------------------------------------------------------------------
// Host launcher
// ---------------------------------------------------------------------------
extern "C" void kernel_launch(void* const* d_in, const int* in_sizes, int n_in,
                              void* d_out, int out_size, void* d_ws, size_t ws_size,
                              hipStream_t stream) {
  (void)in_sizes; (void)n_in; (void)out_size; (void)ws_size;

  const float* x         = (const float*)d_in[0];
  const float* norm_w    = (const float*)d_in[1];
  const float* norm_b    = (const float*)d_in[2];
  const float* x_proj_w  = (const float*)d_in[3];  // (256, 2048) row-major = W(N,K)
  const float* dt_proj_w = (const float*)d_in[4];  // (2048, 128) row-major = W(N,K)
  const float* dt_proj_b = (const float*)d_in[5];
  const float* A_log     = (const float*)d_in[6];
  const float* D_param   = (const float*)d_in[7];
  const float* out_proj_w= (const float*)d_in[8];  // (2048, 2048) row-major = W(N,K)
  float* out = (float*)d_out;

  char* ws = (char*)d_ws;
  size_t off = 0;
  auto alloc = [&](size_t bytes) -> void* {
    void* p = ws + off;
    off += (bytes + 255) & ~(size_t)255;
    return p;
  };
  float*  xn_f32    = (float*) alloc((size_t)BLTOK * DIMD   * 4);
  __bf16* xn_bf16   = (__bf16*)alloc((size_t)BLTOK * DIMD   * 2);
  float*  proj_f32  = (float*) alloc((size_t)BLTOK * 256    * 4);
  __bf16* din_bf16  = (__bf16*)alloc((size_t)BLTOK * DTRANK * 2);
  float*  delta_f32 = (float*) alloc((size_t)BLTOK * DIMD   * 4);
  __bf16* y2_bf16   = (__bf16*)alloc((size_t)BLTOK * DIMD   * 2);
  __bf16* wxp_bf16  = (__bf16*)alloc((size_t)256   * DIMD   * 2);
  __bf16* wdt_bf16  = (__bf16*)alloc((size_t)DIMD  * DTRANK * 2);
  __bf16* wop_bf16  = (__bf16*)alloc((size_t)DIMD  * DIMD   * 2);

  // Weight conversions f32 -> bf16
  f32_to_bf16_kernel<<<256, 256, 0, stream>>>(x_proj_w,  wxp_bf16, (size_t)256  * DIMD);
  f32_to_bf16_kernel<<<256, 256, 0, stream>>>(dt_proj_w, wdt_bf16, (size_t)DIMD * DTRANK);
  f32_to_bf16_kernel<<<1024, 256, 0, stream>>>(out_proj_w, wop_bf16, (size_t)DIMD * DIMD);

  // 1) LayerNorm
  layernorm_kernel<<<BLTOK, 256, 0, stream>>>(x, norm_w, norm_b, xn_f32, xn_bf16);

  // 2) proj = xn @ x_proj_w.T   (M=8192, N=256, K=2048)
  gemm_bf16_wmma<<<dim3(BLTOK / 128, 256 / 64), 256, 0, stream>>>(
      xn_bf16, DIMD, wxp_bf16, DIMD, nullptr, proj_f32, 256, DIMD, 0);

  // 3) bf16 copy of proj[:, :128] for the dt GEMM
  slice_delta_bf16_kernel<<<512, 256, 0, stream>>>(proj_f32, din_bf16);

  // 4) delta = softplus(proj[:, :R] @ dt_proj_w.T + b)  (M=8192, N=2048, K=128)
  gemm_bf16_wmma<<<dim3(BLTOK / 128, DIMD / 64), 256, 0, stream>>>(
      din_bf16, DTRANK, wdt_bf16, DTRANK, dt_proj_b, delta_f32, DIMD, DTRANK, 1);

  // 5) sequential selective scan -> y2 = y + x*D_param (bf16)
  scan_kernel<<<dim3(DIMD / 256, NBATCH), 256, 0, stream>>>(
      xn_f32, proj_f32, delta_f32, A_log, D_param, x, y2_bf16);

  // 6) out = y2 @ out_proj_w.T  (M=8192, N=2048, K=2048)
  gemm_bf16_wmma<<<dim3(BLTOK / 128, DIMD / 64), 256, 0, stream>>>(
      y2_bf16, DIMD, wop_bf16, DIMD, nullptr, out, DIMD, DIMD, 0);
}
